// MiniCharGPTLM_6193342841580
// MI455X (gfx1250) — matrix operand order repaired and checked by reading, not compiled, not run on hardware
//
#include <hip/hip_runtime.h>
#include <hip/hip_bf16.h>
#include <math.h>

typedef __attribute__((ext_vector_type(16))) _Float16 v16h;
typedef __attribute__((ext_vector_type(8)))  float    v8f;

#define H_DIM   512
#define FF_DIM  2048
#define N_HEAD  4
#define N_BLOCK 3
#define N_TOKEN 128
#define BATCH   8
#define SEQ     1024
#define ROWS    (BATCH * SEQ)      // 8192
#define NHH     (N_HEAD * H_DIM)   // 2048
#define BT_LD   40                 // padded LDS transposed-tile stride (attention)

// ---------------------------------------------------------------------------
// Positional scan: pos = cumsum(mask) - 1 along S (Hillis-Steele in LDS)
// ---------------------------------------------------------------------------
__global__ void pos_scan_kernel(const int* __restrict__ mask, float* __restrict__ pos) {
    __shared__ float s[SEQ];
    int b = blockIdx.x, t = threadIdx.x;
    s[t] = (float)mask[b * SEQ + t];
    __syncthreads();
    for (int off = 1; off < SEQ; off <<= 1) {
        float v = (t >= off) ? s[t - off] : 0.0f;
        __syncthreads();
        s[t] += v;
        __syncthreads();
    }
    pos[b * SEQ + t] = s[t] - 1.0f;
}

// ---------------------------------------------------------------------------
// Embedding + sinusoidal positional encoding; writes x in f32 and f16
// ---------------------------------------------------------------------------
__global__ __launch_bounds__(256) void embed_pos_kernel(
    const int* __restrict__ ids, const int* __restrict__ mask,
    const float* __restrict__ emb, const float* __restrict__ pos,
    float* __restrict__ xf, _Float16* __restrict__ xh)
{
    int bs = blockIdx.x;                 // token index 0..8191
    int h  = threadIdx.x * 2;            // even feature slot
    int id = ids[bs];
    float p  = pos[bs];
    float mf = (float)mask[bs];
    float div = __expf((float)h * (-9.210340371976184f / (float)H_DIM));
    float sv  = __sinf(p * div) * mf;
    size_t base = (size_t)bs * H_DIM;
    float x0 = emb[(size_t)id * H_DIM + h]     + sv;
    float x1 = emb[(size_t)id * H_DIM + h + 1] + sv;
    xf[base + h]     = x0;  xf[base + h + 1] = x1;
    xh[base + h]     = (_Float16)x0;
    xh[base + h + 1] = (_Float16)x1;
}

// ---------------------------------------------------------------------------
// Weight conversion kernels: f32 -> f16, emitting B^T layout [N][K] so GEMM
// B-fragments are contiguous 16B global loads (no LDS transpose needed).
// ---------------------------------------------------------------------------
// wq[l] is (N_HEAD, K=h, D); emit bt[(n*512+d)][k]
__global__ void cvt_qkv_kernel(const float* __restrict__ w, _Float16* __restrict__ o) {
    int idx = blockIdx.x * 256 + threadIdx.x;    // over 2048*512 outputs
    int k   = idx & 511;
    int nd  = idx >> 9;          // n*512+d
    int d   = nd & 511;
    int n   = nd >> 9;
    o[idx] = (_Float16)w[(((size_t)n * H_DIM + k) << 9) + d];
}
__global__ void cvt_direct_kernel(const float* __restrict__ w, _Float16* __restrict__ o, int n) {
    int idx = blockIdx.x * 256 + threadIdx.x;
    if (idx < n) o[idx] = (_Float16)w[idx];
}
// w is [R][C]; emit o[C][R]  (o[c*R+r] = w[r*C+c])
__global__ void cvt_trans_kernel(const float* __restrict__ w, _Float16* __restrict__ o, int R, int C) {
    int idx = blockIdx.x * 256 + threadIdx.x;
    if (idx < R * C) {
        int c = idx / R, r = idx % R;
        o[idx] = (_Float16)w[(size_t)r * C + c];
    }
}

// ---------------------------------------------------------------------------
// GEMM fragment set: 2 A-fragments (32 M-rows) + 4 B-fragments (64 N-cols)
// ---------------------------------------------------------------------------
struct GFrag {
    union U { v16h v; float4 f[2]; } a0, a1, b[4];
};

__device__ __forceinline__ void gemm_load(GFrag& g,
    const _Float16* __restrict__ ar0, const _Float16* __restrict__ ar1,
    const _Float16* const* br, int k0)
{
    g.a0.f[0] = *(const float4*)(ar0 + k0);
    g.a0.f[1] = *(const float4*)(ar0 + k0 + 16);
    g.a1.f[0] = *(const float4*)(ar1 + k0);
    g.a1.f[1] = *(const float4*)(ar1 + k0 + 16);
    #pragma unroll
    for (int t = 0; t < 4; ++t) {
        g.b[t].f[0] = *(const float4*)(br[t] + k0);
        g.b[t].f[1] = *(const float4*)(br[t] + k0 + 16);
    }
}

__device__ __forceinline__ void gemm_mma(v8f acc[2][4], const GFrag& g) {
    #pragma unroll
    for (int t = 0; t < 4; ++t) {
        acc[0][t] = __builtin_amdgcn_wmma_f32_16x16x32_f16(
            false, g.a0.v, false, g.b[t].v, (short)0, acc[0][t], false, false);
        acc[1][t] = __builtin_amdgcn_wmma_f32_16x16x32_f16(
            false, g.a1.v, false, g.b[t].v, (short)0, acc[1][t], false, false);
    }
}

// ---------------------------------------------------------------------------
// WMMA GEMM, B pre-transposed: C[M,N] = alpha*A[M,K]*BT[N,K]^T (+bias)(+relu)
// 256 thr = 8 waves (2M x 4N); wave tile 32x64 = 8 accumulators; K-step 32,
// ping-pong double-buffered (loads of one buffer overlap WMMAs of the other).
// mode bits: 1=bias, 2=relu, 4=store f16 to Ch, 8=store f32 to Cf
// ---------------------------------------------------------------------------
__global__ __launch_bounds__(256) void gemm_wmma_kernel(
    const _Float16* __restrict__ A, int lda,
    const _Float16* __restrict__ BT,            // [N][K] row-major
    float* __restrict__ Cf, _Float16* __restrict__ Ch, int ldc,
    const float* __restrict__ bias, float alpha, int mode,
    int M, int N, int K)
{
    const int tid  = threadIdx.x;
    const int lane = tid & 31;
    const int wave = tid >> 5;
    const int mbase = blockIdx.y * 64 + (wave >> 2) * 32;
    const int nbase = blockIdx.x * 256 + (wave & 3) * 64;
    const int r   = lane & 15;
    const int akb = (lane & 16) ? 8 : 0;    // A fragment K base (ISA 16-bit layout)
    const int bkb = (lane & 16) ? 8 : 0;    // B fragment K base

    v8f acc[2][4] = {};
    const _Float16* arow0 = A + (size_t)(mbase + r) * lda + akb;
    const _Float16* arow1 = arow0 + (size_t)16 * lda;
    const _Float16* brow[4];
    #pragma unroll
    for (int t = 0; t < 4; ++t) {
        int gn = nbase + t * 16 + r;
        if (gn >= N) gn = 0;                 // clamp (stores are guarded)
        brow[t] = BT + (size_t)gn * K + bkb;
    }

    // K is a multiple of 64 for all layers (512 / 2048)
    GFrag f0, f1;
    gemm_load(f0, arow0, arow1, brow, 0);
    for (int k0 = 0; k0 < K; k0 += 64) {
        gemm_load(f1, arow0, arow1, brow, k0 + 32);
        gemm_mma(acc, f0);
        int kn = (k0 + 64 < K) ? (k0 + 64) : k0;   // clamped redundant tail load
        gemm_load(f0, arow0, arow1, brow, kn);
        gemm_mma(acc, f1);
    }

    #pragma unroll
    for (int t = 0; t < 4; ++t) {
        int gn = nbase + t * 16 + r;
        if (gn >= N) continue;
        float bv = (mode & 1) ? bias[gn] : 0.0f;
        #pragma unroll
        for (int mi = 0; mi < 2; ++mi) {
            int m0 = mbase + mi * 16 + ((lane & 16) ? 8 : 0);
            #pragma unroll
            for (int i = 0; i < 8; ++i) {
                float v = acc[mi][t][i] * alpha + bv;
                if (mode & 2) v = fmaxf(v, 0.0f);
                size_t idx = (size_t)(m0 + i) * ldc + gn;
                if (mode & 4) Ch[idx] = (_Float16)v;
                if (mode & 8) Cf[idx] = v;
            }
        }
    }
}

// ---------------------------------------------------------------------------
// Causal attention, per (query-tile-of-16, head, batch). 256 thr = 8 waves.
// Q pre-scaled by 1/sqrt(H). Scores f32 LDS, probs f16 LDS, V chunk staged
// transposed in LDS (v_perm packing) so P@V fragments are ds_load_b128 pairs.
// ---------------------------------------------------------------------------
struct QKFrag { union U { v16h v; float4 f[2]; } q, k; };

__device__ __forceinline__ void qk_load(QKFrag& g,
    const _Float16* __restrict__ qrow, const _Float16* __restrict__ krow, int kk)
{
    g.q.f[0] = *(const float4*)(qrow + kk);
    g.q.f[1] = *(const float4*)(qrow + kk + 16);
    g.k.f[0] = *(const float4*)(krow + kk);
    g.k.f[1] = *(const float4*)(krow + kk + 16);
}

__global__ __launch_bounds__(256) void attn_kernel(
    const _Float16* __restrict__ qh, const _Float16* __restrict__ kh,
    const _Float16* __restrict__ vh, _Float16* __restrict__ oh)
{
    extern __shared__ char smem[];
    float*    sc  = (float*)smem;                                   // [16][1024] f32
    _Float16* pp  = (_Float16*)(smem + 16 * SEQ * 4);               // [16][1024] f16
    _Float16* vt  = (_Float16*)(smem + 16 * SEQ * 4 + 16 * SEQ * 2);// [512][BT_LD] f16
    float*    red = (float*)(smem + 16 * SEQ * 4 + 16 * SEQ * 2 + H_DIM * BT_LD * 2);

    const int qb    = blockIdx.x;
    const int head  = blockIdx.y;
    const int b     = blockIdx.z;
    const int qbase = qb * 16;
    const int tid   = threadIdx.x;
    const int lane  = tid & 31;
    const int wave  = tid >> 5;
    const int r     = lane & 15;
    const int akb   = (lane & 16) ? 8 : 0;
    const int bkb   = (lane & 16) ? 8 : 0;
    const int mh    = (lane & 16) ? 8 : 0;
    const size_t ld = NHH;   // 2048

    const _Float16* Q  = qh + (size_t)b * SEQ * ld + (size_t)head * H_DIM;
    const _Float16* Km = kh + (size_t)b * SEQ * ld + (size_t)head * H_DIM;
    const _Float16* V  = vh + (size_t)b * SEQ * ld + (size_t)head * H_DIM;

    // -------- Pass 1: scores = Q K^T (causal tiles only, split over waves)
    for (int kt = wave; kt <= qb; kt += 8) {
        v8f acc = {};
        const _Float16* qrow = Q  + (size_t)(qbase   + r) * ld + akb;
        const _Float16* krow = Km + (size_t)(kt * 16 + r) * ld + bkb;
        QKFrag p0, p1;
        qk_load(p0, qrow, krow, 0);
        for (int kk = 0; kk < H_DIM; kk += 64) {   // ping-pong over d
            qk_load(p1, qrow, krow, kk + 32);
            acc = __builtin_amdgcn_wmma_f32_16x16x32_f16(
                false, p0.q.v, false, p0.k.v, (short)0, acc, false, false);
            int kn = (kk + 64 < H_DIM) ? (kk + 64) : kk;
            qk_load(p0, qrow, krow, kn);
            acc = __builtin_amdgcn_wmma_f32_16x16x32_f16(
                false, p1.q.v, false, p1.k.v, (short)0, acc, false, false);
        }
        int key = kt * 16 + r;
        #pragma unroll
        for (int i = 0; i < 8; ++i) {
            int qrowg = qbase + mh + i;
            float s = (key <= qrowg) ? acc[i] : -__builtin_inff();
            sc[(mh + i) * SEQ + key] = s;
        }
    }
    __syncthreads();

    // -------- Pass 2: softmax over valid columns (16 threads per row)
    const int ncols   = (qb + 1) * 16;
    const int ncols32 = (ncols + 31) & ~31;
    {
        int row = tid >> 4;
        int lc  = tid & 15;
        float mx = -__builtin_inff();
        for (int c = lc; c < ncols32; c += 16) {
            float v = (c < ncols) ? sc[row * SEQ + c] : -__builtin_inff();
            mx = fmaxf(mx, v);
        }
        red[row * 16 + lc] = mx;
        __syncthreads();
        float rowmax = -__builtin_inff();
        #pragma unroll
        for (int j = 0; j < 16; ++j) rowmax = fmaxf(rowmax, red[row * 16 + j]);
        __syncthreads();
        float s = 0.0f;
        for (int c = lc; c < ncols32; c += 16) {
            float v = (c < ncols) ? sc[row * SEQ + c] : -__builtin_inff();
            float e = __expf(v - rowmax);      // exp(-inf)=0 handles mask & pad
            sc[row * SEQ + c] = e;
            s += e;
        }
        red[row * 16 + lc] = s;
        __syncthreads();
        float rs = 0.0f;
        #pragma unroll
        for (int j = 0; j < 16; ++j) rs += red[row * 16 + j];
        float inv = 1.0f / rs;                 // diagonal always valid -> rs>0
        for (int c = lc; c < ncols32; c += 16)
            pp[row * SEQ + c] = (_Float16)(sc[row * SEQ + c] * inv);
    }
    __syncthreads();

    // -------- Pass 3: O = P @ V with cooperative V-chunk transpose staging
    const int nkc = ncols32 >> 5;
    const int db  = wave * 64;
    const int skr = (tid >> 4) << 1;      // 0,2,...,30  (2 key rows)
    const int sns = (tid & 15) << 5;      // 0,32,...,480 (32 d-cols)
    v8f o[4] = {};
    for (int kc = 0; kc < nkc; ++kc) {
        __syncthreads();
        {
            const _Float16* v0 = V + (size_t)(kc * 32 + skr) * ld + sns;
            union { float4 f[4]; unsigned d[8]; } r0, r1;
            r0.f[0] = *(const float4*)(v0);      r0.f[1] = *(const float4*)(v0 + 8);
            r0.f[2] = *(const float4*)(v0 + 16); r0.f[3] = *(const float4*)(v0 + 24);
            r1.f[0] = *(const float4*)(v0 + ld);      r1.f[1] = *(const float4*)(v0 + ld + 8);
            r1.f[2] = *(const float4*)(v0 + ld + 16); r1.f[3] = *(const float4*)(v0 + ld + 24);
            #pragma unroll
            for (int j2 = 0; j2 < 16; ++j2) {
                // pack {row0,row1} halves via v_perm_b32 (s1 = low bytes)
                unsigned lo = __builtin_amdgcn_perm(r1.d[j2], r0.d[j2], 0x05040100u);
                unsigned hi = __builtin_amdgcn_perm(r1.d[j2], r0.d[j2], 0x07060302u);
                *(unsigned*)(&vt[(sns + 2 * j2)     * BT_LD + skr]) = lo;
                *(unsigned*)(&vt[(sns + 2 * j2 + 1) * BT_LD + skr]) = hi;
            }
        }
        __syncthreads();
        union { v16h v; float4 f[2]; } ua;
        const _Float16* prow = pp + r * SEQ + kc * 32 + akb;
        ua.f[0] = *(const float4*)(prow);
        ua.f[1] = *(const float4*)(prow + 16);
        #pragma unroll
        for (int t = 0; t < 4; ++t) {
            union { v16h v; float4 f[2]; } ub;
            const _Float16* bp = &vt[(db + t * 16 + r) * BT_LD + bkb];
            ub.f[0] = *(const float4*)(bp);
            ub.f[1] = *(const float4*)(bp + 16);
            o[t] = __builtin_amdgcn_wmma_f32_16x16x32_f16(
                false, ua.v, false, ub.v, (short)0, o[t], false, false);
        }
    }
    _Float16* O = oh + ((size_t)b * SEQ + qbase) * ld + (size_t)head * H_DIM;
    #pragma unroll
    for (int t = 0; t < 4; ++t) {
        int dc = db + t * 16 + r;
        #pragma unroll
        for (int i = 0; i < 8; ++i)
            O[(size_t)(mh + i) * ld + dc] = (_Float16)o[t][i];
    }
}

// ---------------------------------------------------------------------------
// Residual add + LayerNorm over H=512; writes f32 and f16 copies of x
// ---------------------------------------------------------------------------
__global__ __launch_bounds__(256) void resid_ln_kernel(
    const float* __restrict__ xin, const float* __restrict__ delta,
    const float* __restrict__ g, const float* __restrict__ bta,
    float* __restrict__ xout, _Float16* __restrict__ xhout)
{
    __shared__ float sred[256];
    int row = blockIdx.x;
    int tid = threadIdx.x;
    size_t base = (size_t)row * H_DIM;
    float v0 = xin[base + tid]       + delta[base + tid];
    float v1 = xin[base + tid + 256] + delta[base + tid + 256];
    sred[tid] = v0 + v1;
    __syncthreads();
    for (int s = 128; s > 0; s >>= 1) { if (tid < s) sred[tid] += sred[tid + s]; __syncthreads(); }
    float mu = sred[0] * (1.0f / H_DIM);
    __syncthreads();
    float d0 = v0 - mu, d1 = v1 - mu;
    sred[tid] = d0 * d0 + d1 * d1;
    __syncthreads();
    for (int s = 128; s > 0; s >>= 1) { if (tid < s) sred[tid] += sred[tid + s]; __syncthreads(); }
    float rstd = rsqrtf(sred[0] * (1.0f / H_DIM) + 1e-5f);
    float o0 = d0 * rstd * g[tid]       + bta[tid];
    float o1 = d1 * rstd * g[tid + 256] + bta[tid + 256];
    xout[base + tid]        = o0;
    xout[base + tid + 256]  = o1;
    xhout[base + tid]       = (_Float16)o0;
    xhout[base + tid + 256] = (_Float16)o1;
}

// ---------------------------------------------------------------------------
// Host-side orchestration
// ---------------------------------------------------------------------------
extern "C" void kernel_launch(void* const* d_in, const int* in_sizes, int n_in,
                              void* d_out, int out_size, void* d_ws, size_t ws_size,
                              hipStream_t stream) {
    const int*   ids    = (const int*)  d_in[0];
    const int*   amask  = (const int*)  d_in[1];
    const float* emb    = (const float*)d_in[2];
    const float* wq     = (const float*)d_in[3];
    const float* wk     = (const float*)d_in[4];
    const float* wv     = (const float*)d_in[5];
    const float* wo_w   = (const float*)d_in[6];
    const float* wo_b   = (const float*)d_in[7];
    const float* ln1_g  = (const float*)d_in[8];
    const float* ln1_b  = (const float*)d_in[9];
    const float* ff1_w  = (const float*)d_in[10];
    const float* ff1_b  = (const float*)d_in[11];
    const float* ff2_w  = (const float*)d_in[12];
    const float* ff2_b  = (const float*)d_in[13];
    const float* ln2_g  = (const float*)d_in[14];
    const float* ln2_b  = (const float*)d_in[15];
    const float* lm_w   = (const float*)d_in[16];
    float* out = (float*)d_out;

    char* ws = (char*)d_ws;
    size_t off = 0;
    auto alloc = [&](size_t bytes) -> char* {
        char* p = ws + off;
        off = (off + bytes + 255) & ~(size_t)255;
        return p;
    };

    float*    pos   = (float*)   alloc((size_t)ROWS * 4);
    float*    xf    = (float*)   alloc((size_t)ROWS * H_DIM * 4);
    _Float16* xh    = (_Float16*)alloc((size_t)ROWS * H_DIM * 2);
    _Float16* qall  = (_Float16*)alloc((size_t)ROWS * NHH * 2);
    _Float16* kall  = (_Float16*)alloc((size_t)ROWS * NHH * 2);
    _Float16* vall  = (_Float16*)alloc((size_t)ROWS * NHH * 2);
    _Float16* oall  = (_Float16*)alloc((size_t)ROWS * NHH * 2);
    float*    delta = (float*)   alloc((size_t)ROWS * H_DIM * 4);
    _Float16* ffh   = (_Float16*)alloc((size_t)ROWS * FF_DIM * 2);
    _Float16* wqc   = (_Float16*)alloc((size_t)NHH * H_DIM * 2);   // [2048][512]
    _Float16* wkc   = (_Float16*)alloc((size_t)NHH * H_DIM * 2);
    _Float16* wvc   = (_Float16*)alloc((size_t)NHH * H_DIM * 2);
    _Float16* woc   = (_Float16*)alloc((size_t)H_DIM * NHH * 2);   // [512][2048]
    _Float16* f1c   = (_Float16*)alloc((size_t)FF_DIM * H_DIM * 2);// [2048][512]
    _Float16* f2c   = (_Float16*)alloc((size_t)H_DIM * FF_DIM * 2);// [512][2048]
    _Float16* lmc   = (_Float16*)alloc((size_t)N_TOKEN * H_DIM * 2);// [128][512]
    (void)ws_size; (void)in_sizes; (void)n_in; (void)out_size;

    pos_scan_kernel<<<BATCH, SEQ, 0, stream>>>(amask, pos);
    embed_pos_kernel<<<ROWS, 256, 0, stream>>>(ids, amask, emb, pos, xf, xh);

    const float qscale = 0.044194173824159216f;  // 1/sqrt(512)
    const dim3 g1(NHH / 256, ROWS / 64);      // N=2048
    const dim3 g2(H_DIM / 256, ROWS / 64);    // N=512
    const dim3 g3(1, ROWS / 64);              // N=128
    const dim3 ga(SEQ / 16, N_HEAD, BATCH);
    const size_t attn_lds = 16 * SEQ * 4 + 16 * SEQ * 2 + H_DIM * BT_LD * 2 + 16 * 16 * 4;

    for (int l = 0; l < N_BLOCK; ++l) {
        const size_t wqkv_off = (size_t)l * N_HEAD * H_DIM * H_DIM;
        // QKV weights -> BT [2048][512]
        cvt_qkv_kernel<<<(NHH * H_DIM) / 256, 256, 0, stream>>>(wq + wqkv_off, wqc);
        cvt_qkv_kernel<<<(NHH * H_DIM) / 256, 256, 0, stream>>>(wk + wqkv_off, wkc);
        cvt_qkv_kernel<<<(NHH * H_DIM) / 256, 256, 0, stream>>>(wv + wqkv_off, wvc);
        // wo [2048][512] -> BT [512][2048] (transpose)
        cvt_trans_kernel<<<(NHH * H_DIM) / 256, 256, 0, stream>>>(
            wo_w + (size_t)l * NHH * H_DIM, woc, NHH, H_DIM);
        // ff1 [2048][512] and ff2 [512][2048] are already [N][K]
        cvt_direct_kernel<<<(FF_DIM * H_DIM) / 256, 256, 0, stream>>>(
            ff1_w + (size_t)l * FF_DIM * H_DIM, f1c, FF_DIM * H_DIM);
        cvt_direct_kernel<<<(H_DIM * FF_DIM) / 256, 256, 0, stream>>>(
            ff2_w + (size_t)l * H_DIM * FF_DIM, f2c, H_DIM * FF_DIM);

        // Q (scaled), K, V projections: [8192x512] x [512->2048] -> f16
        gemm_wmma_kernel<<<g1, 256, 0, stream>>>(xh, H_DIM, wqc, nullptr, qall, NHH,
                                                 nullptr, qscale, 4, ROWS, NHH, H_DIM);
        gemm_wmma_kernel<<<g1, 256, 0, stream>>>(xh, H_DIM, wkc, nullptr, kall, NHH,
                                                 nullptr, 1.0f, 4, ROWS, NHH, H_DIM);
        gemm_wmma_kernel<<<g1, 256, 0, stream>>>(xh, H_DIM, wvc, nullptr, vall, NHH,
                                                 nullptr, 1.0f, 4, ROWS, NHH, H_DIM);

        attn_kernel<<<ga, 256, attn_lds, stream>>>(qall, kall, vall, oall);

        // mh = o @ wo + wo_b  -> f32 delta
        gemm_wmma_kernel<<<g2, 256, 0, stream>>>(oall, NHH, woc, delta, nullptr, H_DIM,
                                                 wo_b + (size_t)l * H_DIM, 1.0f, 1 | 8,
                                                 ROWS, H_DIM, NHH);
        resid_ln_kernel<<<ROWS, 256, 0, stream>>>(xf, delta,
                                                  ln1_g + (size_t)l * H_DIM,
                                                  ln1_b + (size_t)l * H_DIM, xf, xh);

        // ff = relu(x @ ff1^T + b1) -> f16
        gemm_wmma_kernel<<<g1, 256, 0, stream>>>(xh, H_DIM, f1c, nullptr, ffh, FF_DIM,
                                                 ff1_b + (size_t)l * FF_DIM, 1.0f, 1 | 2 | 4,
                                                 ROWS, FF_DIM, H_DIM);
        // ff2 -> f32 delta
        gemm_wmma_kernel<<<g2, 256, 0, stream>>>(ffh, FF_DIM, f2c, delta, nullptr, H_DIM,
                                                 ff2_b + (size_t)l * H_DIM, 1.0f, 1 | 8,
                                                 ROWS, H_DIM, FF_DIM);
        resid_ln_kernel<<<ROWS, 256, 0, stream>>>(xf, delta,
                                                  ln2_g + (size_t)l * H_DIM,
                                                  ln2_b + (size_t)l * H_DIM, xf, xh);
    }

    // LM head: lm_w [512][128] -> BT [128][512]; [8192x512]x[512->128] -> f32 out
    cvt_trans_kernel<<<(H_DIM * N_TOKEN) / 256, 256, 0, stream>>>(lm_w, lmc, H_DIM, N_TOKEN);
    gemm_wmma_kernel<<<g3, 256, 0, stream>>>(xh, H_DIM, lmc, out, nullptr, N_TOKEN,
                                             nullptr, 1.0f, 8, ROWS, N_TOKEN, H_DIM);
}
